// SelfMHA_54546084659679
// MI455X (gfx1250) — matrix operand order repaired and checked by reading, not compiled, not run on hardware
//
#include <hip/hip_runtime.h>

// Self-attention forward for MI455X (gfx1250), wave32 + WMMA f16 16x16x32.
// b=2, s=2048, d=1024, heads=16, dh=64.
//
//  - f16 matrix inputs, f32 accumulation (f16 mantissa >> bf16, same WMMA rate)
//  - X pre-converted to f16 so GEMM inner loops are pure load+WMMA (no v_cvt)
//  - 32x64 wave tiles in GEMMs: 8 WMMA per 6 fragment loads
//  - flash attention, transposed math (S^T = K*Q^T, O^T = V^T*P^T), 32 queries
//    per wave so K/V fragments are reused twice; softmax stats are per-lane
//  - 1/sqrt(dh) folded into Q (exact power-of-two scale in f16)
//  - global_prefetch for the next key block

typedef __attribute__((ext_vector_type(16))) _Float16 v16h;
typedef __attribute__((ext_vector_type(8)))  float    v8f;

#define BATCH 2
#define SEQ   2048
#define DIM   1024
#define HEADS 16
#define DH    64
#define D3    3072
#define ROWS  (BATCH * SEQ)   // 4096
#define NEGV  (-1e10f)

// ---------------------------------------------------------------------------
// elementwise fp32 -> f16 convert (vector loads, 4 elems/thread)
// ---------------------------------------------------------------------------
__global__ __launch_bounds__(256)
void k_cvt_f16(const float* __restrict__ in, _Float16* __restrict__ out, int n)
{
    const int i = (blockIdx.x * 256 + threadIdx.x) * 4;
    if (i + 3 < n) {
        const float4 v = *reinterpret_cast<const float4*>(in + i);
        out[i + 0] = (_Float16)v.x;
        out[i + 1] = (_Float16)v.y;
        out[i + 2] = (_Float16)v.z;
        out[i + 3] = (_Float16)v.w;
    }
}

// ---------------------------------------------------------------------------
// fp32 [K][N] -> f16 [N][K] transpose+convert (LDS tiled, coalesced both ways)
// ---------------------------------------------------------------------------
__global__ __launch_bounds__(256)
void k_transpose_cvt(const float* __restrict__ in, _Float16* __restrict__ out,
                     int rowsK, int colsN)
{
    __shared__ _Float16 tile[32][33];
    const int nBase = blockIdx.x * 32;
    const int kBase = blockIdx.y * 32;
    for (int i = threadIdx.y; i < 32; i += 8)
        tile[i][threadIdx.x] =
            (_Float16)in[(size_t)(kBase + i) * colsN + nBase + threadIdx.x];
    __syncthreads();
    for (int i = threadIdx.y; i < 32; i += 8)
        out[(size_t)(nBase + i) * rowsK + kBase + threadIdx.x] = tile[threadIdx.x][i];
}

// ---------------------------------------------------------------------------
// QKV projection: qkv[4096,3072] = Xh[4096,1024](f16) @ Wqkv + b
// Wave tile 32x64: 8 WMMA per K-step of 32.
// Writes Q (pre-scaled by 1/8), K head-major [b][h][s][64] and V transposed
// [b][h][64][s], all f16, feature f = dd*16 + h.
// ---------------------------------------------------------------------------
__global__ __launch_bounds__(128)
void k_qkv_gemm(const _Float16* __restrict__ Xh, const _Float16* __restrict__ Wt,
                const float* __restrict__ bias,
                _Float16* __restrict__ Qb, _Float16* __restrict__ Kb,
                _Float16* __restrict__ Vt)
{
    const int lane = threadIdx.x & 31;
    const int wave = threadIdx.x >> 5;
    const int tileId = blockIdx.x * 4 + wave;
    const int numN = D3 / 64;                 // 48
    const int m0 = (tileId / numN) * 32;
    const int n0 = (tileId % numN) * 64;
    const int lr = lane & 15, kg = lane >> 4;

    v8f acc[2][4] = {};
    for (int kk = 0; kk < DIM; kk += 32) {
        v16h a[2], bf[4];
#pragma unroll
        for (int i = 0; i < 2; ++i) {
            const _Float16* ap = Xh + (size_t)(m0 + i * 16 + lr) * DIM + kk;
#pragma unroll
            for (int e = 0; e < 8; ++e) {
                a[i][e]     = ap[kg * 16 + e];
                a[i][8 + e] = ap[kg * 16 + 8 + e];
            }
        }
#pragma unroll
        for (int j = 0; j < 4; ++j) {
            const _Float16* bp = Wt + (size_t)(n0 + j * 16 + lr) * DIM + kk + kg * 16;
#pragma unroll
            for (int e = 0; e < 16; ++e) bf[j][e] = bp[e];
        }
#pragma unroll
        for (int i = 0; i < 2; ++i)
#pragma unroll
            for (int j = 0; j < 4; ++j)
                acc[i][j] = __builtin_amdgcn_wmma_f32_16x16x32_f16(
                    false, a[i], false, bf[j], (short)0, acc[i][j], false, false);
    }

    // Epilogue: bias add, convert, de-interleave heads (feature f = dd*16 + h)
#pragma unroll
    for (int j = 0; j < 4; ++j) {
        const int c = n0 + j * 16 + lr;
        const float bv = bias[c];
        const int seg = c >> 10;        // 0=Q 1=K 2=V
        const int f = c & 1023;
        const int h = f & 15, dd = f >> 4;
        const float scl = (seg == 0) ? 0.125f : 1.0f;   // fold 1/sqrt(64) into Q
#pragma unroll
        for (int i = 0; i < 2; ++i) {
#pragma unroll
            for (int e = 0; e < 8; ++e) {
                const int r = m0 + i * 16 + kg * 8 + e;   // token row
                const int bb = r >> 11, s = r & 2047;
                const _Float16 val = (_Float16)((acc[i][j][e] + bv) * scl);
                const size_t bh = (size_t)(bb * 16 + h);
                if (seg == 0)      Qb[(bh * SEQ + s) * DH + dd] = val;
                else if (seg == 1) Kb[(bh * SEQ + s) * DH + dd] = val;
                else               Vt[(bh * DH + dd) * SEQ + s] = val;
            }
        }
    }
}

// ---------------------------------------------------------------------------
// Flash attention, transposed formulation. One wave = (b, h, 32-query block):
// two 16-query tiles share every K and V fragment.
//   S^T[key,q] = K_tile(16x64) @ Q^T(64x16)   -> C frag: lane=q col, elem=key
//   online softmax per lane (+1 shfl_xor(16) merge of the two key half-rows)
//   O^T[dh,q] += V^T(16x32) @ P^T(32x16)
// Writes Atn[b][s][dd*16+h] f16.
// ---------------------------------------------------------------------------
__global__ __launch_bounds__(128)
void k_flash_attn(const _Float16* __restrict__ Qb, const _Float16* __restrict__ Kb,
                  const _Float16* __restrict__ Vt, const unsigned char* __restrict__ mask,
                  _Float16* __restrict__ Atn)
{
    const int lane = threadIdx.x & 31;
    const int wave = threadIdx.x >> 5;
    const int unit = blockIdx.x * 4 + wave;   // 0..2047
    const int qB = unit & 63;                 // 64 query blocks of 32
    const int bh = unit >> 6;                 // b*16 + h
    const int b = bh >> 4, h = bh & 15;
    const int q0 = qB * 32;
    const int lr = lane & 15, kg = lane >> 4;

    const _Float16* Qh = Qb + (size_t)bh * SEQ * DH;
    const _Float16* Kh = Kb + (size_t)bh * SEQ * DH;
    const _Float16* Vh = Vt + (size_t)bh * DH * SEQ;
    const unsigned char* mrow[2] = {
        mask + ((size_t)b * SEQ + (q0 + lr)) * SEQ,
        mask + ((size_t)b * SEQ + (q0 + 16 + lr)) * SEQ
    };

    // Q^T B-fragments (Q already scaled by 1/8): lane = q col, elem j -> dh
    v16h qfr[2][2];
#pragma unroll
    for (int u = 0; u < 2; ++u) {
        const _Float16* qp = Qh + (size_t)(q0 + u * 16 + lr) * DH + kg * 16;
#pragma unroll
        for (int e = 0; e < 16; ++e) { qfr[u][0][e] = qp[e]; qfr[u][1][e] = qp[32 + e]; }
    }

    const v8f zacc = {};
    v8f o[2][4] = {};
    float m_i[2] = { -3.0e38f, -3.0e38f };
    float l_i[2] = { 0.0f, 0.0f };

    for (int kb = 0; kb < SEQ; kb += 32) {
        if (kb + 32 < SEQ) {   // prefetch next key block (global_prefetch_b8)
            __builtin_prefetch(Kh + (size_t)(kb + 32 + lr) * DH, 0, 3);
            __builtin_prefetch(Vh + (size_t)lr * SEQ + kb + 32, 0, 3);
        }
        // ---- K A-fragments for two 16-key tiles (shared by both q tiles) --
        v16h ka[2][2];
#pragma unroll
        for (int t = 0; t < 2; ++t) {
            const _Float16* kp = Kh + (size_t)(kb + t * 16 + lr) * DH;
#pragma unroll
            for (int e = 0; e < 8; ++e) {
                ka[t][0][e] = kp[kg * 16 + e];      ka[t][0][8 + e] = kp[kg * 16 + 8 + e];
                ka[t][1][e] = kp[32 + kg * 16 + e]; ka[t][1][8 + e] = kp[40 + kg * 16 + e];
            }
        }
        // ---- scores^T -----------------------------------------------------
        v8f st[2][2];
#pragma unroll
        for (int u = 0; u < 2; ++u)
#pragma unroll
            for (int t = 0; t < 2; ++t) {
                v8f s = __builtin_amdgcn_wmma_f32_16x16x32_f16(
                    false, ka[t][0], false, qfr[u][0], (short)0, zacc, false, false);
                s = __builtin_amdgcn_wmma_f32_16x16x32_f16(
                    false, ka[t][1], false, qfr[u][1], (short)0, s, false, false);
                st[u][t] = s;
            }
        // ---- mask (elem e of tile t is key kb + t*16 + kg*8 + e) ----------
#pragma unroll
        for (int u = 0; u < 2; ++u)
#pragma unroll
            for (int t = 0; t < 2; ++t) {
                const unsigned long long mv = *reinterpret_cast<const unsigned long long*>(
                    mrow[u] + kb + t * 16 + kg * 8);
#pragma unroll
                for (int e = 0; e < 8; ++e)
                    if (!((mv >> (8 * e)) & 0xffull)) st[u][t][e] = NEGV;
            }
        // ---- online softmax + P^T fragments -------------------------------
        v16h pb[2];
#pragma unroll
        for (int u = 0; u < 2; ++u) {
            float bm = st[u][0][0];
#pragma unroll
            for (int e = 1; e < 8; ++e) bm = fmaxf(bm, st[u][0][e]);
#pragma unroll
            for (int e = 0; e < 8; ++e) bm = fmaxf(bm, st[u][1][e]);
            bm = fmaxf(bm, __shfl_xor(bm, 16, 32));
            const float m_new = fmaxf(m_i[u], bm);
            float p0[8], p1[8], rs = 0.0f;
#pragma unroll
            for (int e = 0; e < 8; ++e) { p0[e] = __expf(st[u][0][e] - m_new); rs += p0[e]; }
#pragma unroll
            for (int e = 0; e < 8; ++e) { p1[e] = __expf(st[u][1][e] - m_new); rs += p1[e]; }
            rs += __shfl_xor(rs, 16, 32);
            const float corr = __expf(m_i[u] - m_new);
            l_i[u] = l_i[u] * corr + rs;
            m_i[u] = m_new;
#pragma unroll
            for (int d = 0; d < 4; ++d)
#pragma unroll
                for (int e = 0; e < 8; ++e) o[u][d][e] *= corr;
            // P^T B-fragment (32 keys x 16 q) via shfl_xor(16)
            float x0[8], x1[8];
#pragma unroll
            for (int e = 0; e < 8; ++e) {
                x0[e] = __shfl_xor(p0[e], 16, 32);
                x1[e] = __shfl_xor(p1[e], 16, 32);
            }
#pragma unroll
            for (int e = 0; e < 8; ++e) {
                pb[u][e]     = (_Float16)((lane < 16) ? p0[e] : x1[e]);
                pb[u][8 + e] = (_Float16)((lane < 16) ? x0[e] : p1[e]);
            }
        }
        // ---- O^T += V^T @ P^T (V fragments shared by both q tiles) --------
#pragma unroll
        for (int d = 0; d < 4; ++d) {
            v16h va;
            const _Float16* vp = Vh + (size_t)(d * 16 + lr) * SEQ + kb;
#pragma unroll
            for (int e = 0; e < 8; ++e) {
                va[e] = vp[kg * 16 + e];
                va[8 + e] = vp[kg * 16 + 8 + e];
            }
#pragma unroll
            for (int u = 0; u < 2; ++u)
                o[u][d] = __builtin_amdgcn_wmma_f32_16x16x32_f16(
                    false, va, false, pb[u], (short)0, o[u][d], false, false);
        }
    }

    // ---- normalize + store: Atn[b][s][dd*16+h] ----------------------------
#pragma unroll
    for (int u = 0; u < 2; ++u) {
        const float inv = 1.0f / l_i[u];
        const int s = q0 + u * 16 + lr;
        _Float16* outp = Atn + ((size_t)b * SEQ + s) * DIM + h;
#pragma unroll
        for (int d = 0; d < 4; ++d)
#pragma unroll
            for (int e = 0; e < 8; ++e) {
                const int dd = d * 16 + kg * 8 + e;
                outp[dd * 16] = (_Float16)(o[u][d][e] * inv);
            }
    }
}

// ---------------------------------------------------------------------------
// Output projection: Y[4096,1024] = Atn(f16) @ Wout + b_out (fp32 out)
// Wave tile 32x64.
// ---------------------------------------------------------------------------
__global__ __launch_bounds__(128)
void k_out_gemm(const _Float16* __restrict__ A, const _Float16* __restrict__ Wt,
                const float* __restrict__ bias, float* __restrict__ Y)
{
    const int lane = threadIdx.x & 31;
    const int wave = threadIdx.x >> 5;
    const int tileId = blockIdx.x * 4 + wave;
    const int numN = DIM / 64;                // 16
    const int m0 = (tileId / numN) * 32;
    const int n0 = (tileId % numN) * 64;
    const int lr = lane & 15, kg = lane >> 4;

    v8f acc[2][4] = {};
    for (int kk = 0; kk < DIM; kk += 32) {
        v16h a[2], bf[4];
#pragma unroll
        for (int i = 0; i < 2; ++i) {
            const _Float16* ap = A + (size_t)(m0 + i * 16 + lr) * DIM + kk;
#pragma unroll
            for (int e = 0; e < 8; ++e) {
                a[i][e]     = ap[kg * 16 + e];
                a[i][8 + e] = ap[kg * 16 + 8 + e];
            }
        }
#pragma unroll
        for (int j = 0; j < 4; ++j) {
            const _Float16* bp = Wt + (size_t)(n0 + j * 16 + lr) * DIM + kk + kg * 16;
#pragma unroll
            for (int e = 0; e < 16; ++e) bf[j][e] = bp[e];
        }
#pragma unroll
        for (int i = 0; i < 2; ++i)
#pragma unroll
            for (int j = 0; j < 4; ++j)
                acc[i][j] = __builtin_amdgcn_wmma_f32_16x16x32_f16(
                    false, a[i], false, bf[j], (short)0, acc[i][j], false, false);
    }
#pragma unroll
    for (int j = 0; j < 4; ++j) {
        const int c = n0 + j * 16 + lr;
        const float bv = bias[c];
#pragma unroll
        for (int i = 0; i < 2; ++i)
#pragma unroll
            for (int e = 0; e < 8; ++e) {
                const int r = m0 + i * 16 + kg * 8 + e;
                Y[(size_t)r * DIM + c] = acc[i][j][e] + bv;
            }
    }
}

// ---------------------------------------------------------------------------
extern "C" void kernel_launch(void* const* d_in, const int* in_sizes, int n_in,
                              void* d_out, int out_size, void* d_ws, size_t ws_size,
                              hipStream_t stream)
{
    const float*         xs   = (const float*)d_in[0];
    const unsigned char* mask = (const unsigned char*)d_in[1];   // jnp bool = 1 byte
    const float*         Wqkv = (const float*)d_in[2];
    const float*         bqkv = (const float*)d_in[3];
    const float*         Wout = (const float*)d_in[4];
    const float*         bout = (const float*)d_in[5];
    float*               Y    = (float*)d_out;

    // Workspace carve-up (all f16), total ~48 MB
    _Float16* WqkvT = (_Float16*)d_ws;                      // [3072][1024]
    _Float16* WoutT = WqkvT + (size_t)D3 * DIM;             // [1024][1024]
    _Float16* Qb    = WoutT + (size_t)DIM * DIM;            // [b][h][s][64]
    _Float16* Kb    = Qb + (size_t)BATCH * HEADS * SEQ * DH;
    _Float16* Vt    = Kb + (size_t)BATCH * HEADS * SEQ * DH;// [b][h][64][s]
    _Float16* Atn   = Vt + (size_t)BATCH * HEADS * DH * SEQ;// [b][s][1024]
    _Float16* Xh    = Atn + (size_t)ROWS * DIM;             // [4096][1024]

    dim3 tb(32, 8);
    k_cvt_f16<<<(ROWS * DIM) / (256 * 4), 256, 0, stream>>>(xs, Xh, ROWS * DIM);
    k_transpose_cvt<<<dim3(D3 / 32, DIM / 32), tb, 0, stream>>>(Wqkv, WqkvT, DIM, D3);
    k_transpose_cvt<<<dim3(DIM / 32, DIM / 32), tb, 0, stream>>>(Wout, WoutT, DIM, DIM);

    // 128 x 48 tiles of 32x64, 4 waves/block
    k_qkv_gemm<<<(ROWS / 32) * (D3 / 64) / 4, 128, 0, stream>>>(Xh, WqkvT, bqkv, Qb, Kb, Vt);

    // b*h*64 query blocks of 32, 4 waves/block
    k_flash_attn<<<(BATCH * HEADS * (SEQ / 32)) / 4, 128, 0, stream>>>(Qb, Kb, Vt, mask, Atn);

    // 128 x 16 tiles of 32x64, 4 waves/block
    k_out_gemm<<<(ROWS / 32) * (DIM / 64) / 4, 128, 0, stream>>>(Atn, WoutT, bout, Y);
}
